// StaticDecEmbeddingLSTM_35897336660148
// MI455X (gfx1250) — matrix operand order repaired and checked by reading, not compiled, not run on hardware
//
#include <hip/hip_runtime.h>

// Problem constants
#define NB    4096      // batch
#define DD    512       // embed dim (K of main GEMM)
#define NFE   32        // numerical features
#define NCE   8         // categorical features
#define HH    64        // hidden
#define KK    16        // categories
#define OUTW  160       // NF + NC*K
#define EMB_ELEMS (NB * OUTW)

#define XB_ELEMS   (NB * DD)              // 2097152
#define WIH_ELEMS  (NFE * 4 * HH * DD)    // 4194304
#define WC1_ELEMS  (NCE * HH * DD)        // 262144
#define WC2_ELEMS  (NCE * KK * HH)        // 8192

typedef __attribute__((ext_vector_type(8)))  __bf16 v8bf;
typedef __attribute__((ext_vector_type(16))) __bf16 v16bf;
typedef __attribute__((ext_vector_type(8)))  float  v8f;

__device__ __forceinline__ unsigned short f32_to_bf16_rne(float f) {
    unsigned int u = __float_as_uint(f);
    unsigned int r = u + 0x7FFFu + ((u >> 16) & 1u);
    return (unsigned short)(r >> 16);
}

__device__ __forceinline__ float sigmoidf_(float x) {
    return 1.0f / (1.0f + __expf(-x));
}
__device__ __forceinline__ float gelu_exact_(float x) {
    return 0.5f * x * (1.0f + erff(x * 0.70710678118654752f));
}

// Build a v16bf WMMA operand from two contiguous 16-byte bf16 chunks.
// Elements 0..7 <- p0[0..7], elements 8..15 <- p1[0..7] (matches VGPR pair layout).
__device__ __forceinline__ v16bf make_frag(const unsigned short* p0, const unsigned short* p1) {
    v8bf lo = *reinterpret_cast<const v8bf*>(p0);
    v8bf hi = *reinterpret_cast<const v8bf*>(p1);
    v16bf r;
#pragma unroll
    for (int i = 0; i < 8; ++i) { r[i] = lo[i]; r[i + 8] = hi[i]; }
    return r;
}

// ---------------- prep: fp32 -> bf16 conversions + idx arrays -----------------
__global__ void __launch_bounds__(256)
prep_kernel(const float* __restrict__ x, const float* __restrict__ wih,
            const float* __restrict__ wc1, const float* __restrict__ wc2,
            unsigned short* __restrict__ xb, unsigned short* __restrict__ wihb,
            unsigned short* __restrict__ wc1b, unsigned short* __restrict__ wc2b,
            float* __restrict__ out) {
    size_t i = (size_t)blockIdx.x * 256 + threadIdx.x;
    if (i < WIH_ELEMS) wihb[i] = f32_to_bf16_rne(wih[i]);
    if (i < XB_ELEMS)  xb[i]   = f32_to_bf16_rne(x[i]);
    if (i < WC1_ELEMS) wc1b[i] = f32_to_bf16_rne(wc1[i]);
    if (i < WC2_ELEMS) wc2b[i] = f32_to_bf16_rne(wc2[i]);
    if (i < 40) {
        int width = (i < 32) ? 1 : 16;
        int end   = (i < 32) ? (int)(i + 1) : (32 + ((int)i - 31) * 16);
        int* ip = (int*)(out + EMB_ELEMS);
        ip[i]      = end - width;  // idx_start
        ip[40 + i] = end;          // idx_end
    }
}

// ---------------- fused main kernel -----------------
// blocks 0..1023   : numerical  (f = b>>5, mtile = b&31), 128 rows x (i,g,o gates)
// blocks 1024..1279: categorical (c = (b-1024)>>5, mtile), 128 rows x 64 + 64->16
//
// Weight tiles are staged through LDS (double-buffered) so each workgroup pulls
// the W tile from global once per K-step instead of once per wave.
__global__ void __launch_bounds__(256)
embed_kernel(const unsigned short* __restrict__ xb,
             const unsigned short* __restrict__ wihb,
             const unsigned short* __restrict__ wc1b,
             const unsigned short* __restrict__ wc2b,
             const float* __restrict__ b_ih,  const float* __restrict__ b_hh,
             const float* __restrict__ wout,  const float* __restrict__ bout,
             const float* __restrict__ bc1,   const float* __restrict__ bc2,
             float* __restrict__ out) {
    // 24 KB union:
    //  numerical: wbuf[2] of 192x32 bf16 (6144 elems each)
    //  categorical: cbuf[2] of 64x32 bf16 (2048 each) + hlds 128x64 bf16 (8192) at +4096
    __shared__ __attribute__((aligned(16))) unsigned short smem[12288];

    const int tid  = threadIdx.x;
    const int w    = tid >> 5;         // wave 0..7
    const int lane = tid & 31;
    const int ln   = lane & 15;
    const bool hiL = lane >= 16;
    const int aoff = hiL ? 8 : 0;
    const int boff = hiL ? 16 : 0;
    const int bid  = blockIdx.x;

    if (bid < 1024) {
        // ------------------ numerical LSTM features ------------------
        const int f  = bid >> 5;
        const int m0 = (bid & 31) * 128;
        const int mrow = m0 + w * 16;

        v8f acc[12] = {};   // tiles 0..3 = i, 4..7 = g, 8..11 = o (16 cols each)

        const unsigned short* aRow = xb + (size_t)(mrow + ln) * DD;
        const unsigned short* wBase = wihb + (size_t)f * 256 * DD;

        for (int s = 0; s < 16; ++s) {
            const int k0 = s * 32;
            unsigned short* buf = smem + (s & 1) * 6144;
            // cooperative fill: 768 16B chunks, 3 per thread (rows 0..191 of i|g|o)
#pragma unroll
            for (int t = 0; t < 3; ++t) {
                const int ch   = tid + t * 256;
                const int row  = ch >> 2;           // 0..191
                const int col8 = (ch & 3) * 8;
                const int gate = row >> 6;
                const int goff = (gate == 0) ? 0 : ((gate == 1) ? 128 : 192);
                const unsigned short* src =
                    wBase + (size_t)(goff + (row & 63)) * DD + k0 + col8;
                *reinterpret_cast<v8bf*>(buf + row * 32 + col8) =
                    *reinterpret_cast<const v8bf*>(src);
                if (k0 + 32 < DD) __builtin_prefetch(src + 32, 0, 3);
            }
            __syncthreads();

            v16bf afrag = make_frag(aRow + k0 + aoff, aRow + k0 + aoff + 16);
#pragma unroll
            for (int tt = 0; tt < 12; ++tt) {
                const unsigned short* bRow = buf + (tt * 16 + ln) * 32;
                v16bf bfrag = make_frag(bRow + aoff, bRow + aoff + 16);
                acc[tt] = __builtin_amdgcn_wmma_f32_16x16x32_bf16(
                    false, afrag, false, bfrag, (short)0, acc[tt], false, false);
            }
        }

        // biases (col-dependent only) and output weights, per tile
        float bi[4], bg[4], bo[4], wo[4];
#pragma unroll
        for (int t = 0; t < 4; ++t) {
            const int cI = f * 256 + t * 16 + ln;
            bi[t] = b_ih[cI]       + b_hh[cI];
            bg[t] = b_ih[cI + 128] + b_hh[cI + 128];
            bo[t] = b_ih[cI + 192] + b_hh[cI + 192];
            wo[t] = wout[f * 64 + t * 16 + ln];
        }

        float red[8];
#pragma unroll
        for (int r = 0; r < 8; ++r) {
            float s = 0.0f;
#pragma unroll
            for (int t = 0; t < 4; ++t) {
                const float iv = acc[t][r]     + bi[t];
                const float gv = acc[4 + t][r] + bg[t];
                const float ov = acc[8 + t][r] + bo[t];
                const float cc = sigmoidf_(iv) * tanhf(gv);   // f-gate * c0 == 0
                float h = sigmoidf_(ov) * tanhf(cc);
                h = gelu_exact_(h);
                s += h * wo[t];
            }
#pragma unroll
            for (int m = 1; m < 16; m <<= 1) s += __shfl_xor(s, m, 32);
            red[r] = s;
        }
        if (ln == 0) {
            const float bb = bout[f];
            const int rowbase = mrow + (hiL ? 8 : 0);
#pragma unroll
            for (int r = 0; r < 8; ++r)
                out[(size_t)(rowbase + r) * OUTW + f] = red[r] + bb;
        }
    } else {
        // ------------------ categorical MLP features ------------------
        const int cb = bid - 1024;
        const int c  = cb >> 5;
        const int m0 = (cb & 31) * 128;
        const int mrow = m0 + w * 16;
        unsigned short* hlds = smem + 4096;   // 128 x 64 bf16

        v8f acc[4] = {};
        const unsigned short* aRow = xb + (size_t)(mrow + ln) * DD;
        const unsigned short* wBase = wc1b + (size_t)c * 64 * DD;

        for (int s = 0; s < 16; ++s) {
            const int k0 = s * 32;
            unsigned short* buf = smem + (s & 1) * 2048;
            // cooperative fill: 256 16B chunks, 1 per thread (rows 0..63)
            {
                const int row  = tid >> 2;          // 0..63
                const int col8 = (tid & 3) * 8;
                const unsigned short* src = wBase + (size_t)row * DD + k0 + col8;
                *reinterpret_cast<v8bf*>(buf + row * 32 + col8) =
                    *reinterpret_cast<const v8bf*>(src);
                if (k0 + 32 < DD) __builtin_prefetch(src + 32, 0, 3);
            }
            __syncthreads();

            v16bf afrag = make_frag(aRow + k0 + aoff, aRow + k0 + aoff + 16);
#pragma unroll
            for (int t = 0; t < 4; ++t) {
                const unsigned short* bRow = buf + (t * 16 + ln) * 32;
                v16bf bfrag = make_frag(bRow + aoff, bRow + aoff + 16);
                acc[t] = __builtin_amdgcn_wmma_f32_16x16x32_bf16(
                    false, afrag, false, bfrag, (short)0, acc[t], false, false);
            }
        }

        // bias + exact GELU, stage bf16 h-tile [128 x 64] into LDS
#pragma unroll
        for (int t = 0; t < 4; ++t) {
            const float b1 = bc1[c * 64 + t * 16 + ln];
#pragma unroll
            for (int r = 0; r < 8; ++r) {
                const float h = gelu_exact_(acc[t][r] + b1);
                const int row = w * 16 + r + (hiL ? 8 : 0);
                hlds[row * 64 + t * 16 + ln] = f32_to_bf16_rne(h);
            }
        }
        __syncthreads();

        // second layer: [16x64] x [64x16] via two WMMA K-steps
        v8f oacc = {};
        const unsigned short* aL = hlds + (size_t)(w * 16 + ln) * 64;
        const unsigned short* bR = wc2b + (size_t)(c * 16 + ln) * 64;
#pragma unroll
        for (int k0 = 0; k0 < 64; k0 += 32) {
            v16bf afrag = make_frag(aL + k0 + aoff, aL + k0 + aoff + 16);
            v16bf bfrag = make_frag(bR + k0 + aoff, bR + k0 + aoff + 16);
            oacc = __builtin_amdgcn_wmma_f32_16x16x32_bf16(
                false, afrag, false, bfrag, (short)0, oacc, false, false);
        }
        const int kout = ln;
        const float b2 = bc2[c * 16 + kout];
        const int rowbase = mrow + (hiL ? 8 : 0);
#pragma unroll
        for (int r = 0; r < 8; ++r)
            out[(size_t)(rowbase + r) * OUTW + 32 + c * 16 + kout] = oacc[r] + b2;
    }
}

extern "C" void kernel_launch(void* const* d_in, const int* in_sizes, int n_in,
                              void* d_out, int out_size, void* d_ws, size_t ws_size,
                              hipStream_t stream) {
    const float* x     = (const float*)d_in[0];
    const float* W_ih  = (const float*)d_in[1];
    const float* b_ih  = (const float*)d_in[2];
    const float* b_hh  = (const float*)d_in[3];
    const float* W_out = (const float*)d_in[4];
    const float* b_out = (const float*)d_in[5];
    const float* W_c1  = (const float*)d_in[6];
    const float* b_c1  = (const float*)d_in[7];
    const float* W_c2  = (const float*)d_in[8];
    const float* b_c2  = (const float*)d_in[9];
    float* out = (float*)d_out;

    unsigned short* xb   = (unsigned short*)d_ws;
    unsigned short* wihb = xb   + XB_ELEMS;
    unsigned short* wc1b = wihb + WIH_ELEMS;
    unsigned short* wc2b = wc1b + WC1_ELEMS;

    prep_kernel<<<WIH_ELEMS / 256, 256, 0, stream>>>(x, W_ih, W_c1, W_c2,
                                                     xb, wihb, wc1b, wc2b, out);
    embed_kernel<<<1280, 256, 0, stream>>>(xb, wihb, wc1b, wc2b,
                                           b_ih, b_hh, W_out, b_out,
                                           b_c1, b_c2, out);
}
